// Controller_89069031785057
// MI455X (gfx1250) — matrix operand order repaired and checked by reading, not compiled, not run on hardware
//
#include <hip/hip_runtime.h>
#include <math.h>

// Problem constants (from reference)
#define BQ   2
#define LQ   24
#define DQ   256
#define DTQ  1024
#define NLQ  12
#define NHQ  4
#define HDQ  64
#define FFNQ 1024
#define CQ   2
#define MQ   48   // B*L rows

typedef __attribute__((ext_vector_type(16))) __bf16 v16bf;
typedef __attribute__((ext_vector_type(8)))  float  v8f;

// Pack 16 floats -> 16 bf16 lanes of a WMMA fragment via HW cvt (v_cvt_pk_bf16_f32)
__device__ __forceinline__ v16bf pack16(float4 f0, float4 f1, float4 f2, float4 f3) {
  v16bf a;
  a[0]=(__bf16)f0.x;  a[1]=(__bf16)f0.y;  a[2]=(__bf16)f0.z;  a[3]=(__bf16)f0.w;
  a[4]=(__bf16)f1.x;  a[5]=(__bf16)f1.y;  a[6]=(__bf16)f1.z;  a[7]=(__bf16)f1.w;
  a[8]=(__bf16)f2.x;  a[9]=(__bf16)f2.y;  a[10]=(__bf16)f2.z; a[11]=(__bf16)f2.w;
  a[12]=(__bf16)f3.x; a[13]=(__bf16)f3.y; a[14]=(__bf16)f3.z; a[15]=(__bf16)f3.w;
  return a;
}

// A fragment: 16x32 bf16 tile, row = lane&15 of X[M,K] (row-major), K-chunk at k0.
// Lanes 0-15: elems 0-7 -> K=k0+0..7, elems 8-15 -> K=k0+16..23
// Lanes16-31: elems 0-7 -> K=k0+8..15, elems 8-15 -> K=k0+24..31
__device__ __forceinline__ v16bf load_frag_a(const float* __restrict__ rowbase, int k0, int g) {
  const float* p = rowbase + k0 + (g ? 8 : 0);
  return pack16(*(const float4*)(p),      *(const float4*)(p + 4),
                *(const float4*)(p + 16), *(const float4*)(p + 20));
}

// B fragment: 32x16 bf16 tile = W[n0+(lane&15), k0..k0+31] (W row-major [N,K]).
// Lanes 0-15: elems 0-7 -> K=k0+0..7, elems 8-15 -> K=k0+16..23; lanes 16-31: K=k0+8..15, k0+24..31.
__device__ __forceinline__ v16bf load_frag_b(const float* __restrict__ rowbase, int k0, int g) {
  const float* p = rowbase + k0 + (g ? 8 : 0);
  return pack16(*(const float4*)(p),      *(const float4*)(p + 4),
                *(const float4*)(p + 16), *(const float4*)(p + 20));
}

// out[M,N] = X[M,K] @ W[N,K]^T (+bias[n]) (+resid[m,n]).
// One wave computes one 16-row M-tile x two adjacent 16-col N-tiles:
// the A fragment is loaded once per k-step and feeds 2 WMMAs.
template <bool HAS_BIAS, bool HAS_RESID>
__global__ void gemm_wmma_kernel(const float* __restrict__ X, const float* __restrict__ W,
                                 const float* __restrict__ bias, const float* resid,
                                 float* __restrict__ out, int N, int K) {
  const int groups = N >> 5;                   // pairs of 16-col tiles
  const int total  = 3 * groups;               // M=48 -> 3 M-tiles
  const int wave   = blockIdx.x * (blockDim.x >> 5) + (threadIdx.x >> 5);
  if (wave >= total) return;                   // wave-uniform: EXEC stays all-ones
  const int lane = threadIdx.x & 31;
  const int g    = lane >> 4;
  const int rr   = lane & 15;
  const int mt   = wave / groups;
  const int ng   = wave % groups;
  const float* xrow  = X + (size_t)(mt * 16 + rr) * K;
  const float* wrow0 = W + (size_t)(ng * 32 + rr) * K;
  const float* wrow1 = wrow0 + (size_t)16 * K;
  v8f acc0 = {}, acc1 = {};
  for (int k0 = 0; k0 < K; k0 += 32) {
    __builtin_prefetch(wrow0 + k0 + 64, 0, 0); // global_prefetch_b8 on weight stream
    v16bf a  = load_frag_a(xrow, k0, g);
    v16bf b0 = load_frag_b(wrow0, k0, g);
    v16bf b1 = load_frag_b(wrow1, k0, g);
    acc0 = __builtin_amdgcn_wmma_f32_16x16x32_bf16(false, a, false, b0, (short)0, acc0, false, false);
    acc1 = __builtin_amdgcn_wmma_f32_16x16x32_bf16(false, a, false, b1, (short)0, acc1, false, false);
  }
  const int n0 = ng * 32 + rr;
  const int n1 = n0 + 16;
  float bv0 = 0.0f, bv1 = 0.0f;
  if (HAS_BIAS) { bv0 = bias[n0]; bv1 = bias[n1]; }
#pragma unroll
  for (int r = 0; r < 8; ++r) {
    const int m = (mt << 4) + r + (g << 3);
    float v0 = acc0[r] + bv0;
    float v1 = acc1[r] + bv1;
    if (HAS_RESID) {
      v0 += resid[(size_t)m * N + n0];
      v1 += resid[(size_t)m * N + n1];
    }
    out[(size_t)m * N + n0] = v0;
    out[(size_t)m * N + n1] = v1;
  }
}

// hidden[M,FFN] = silu(X@W12[0:FFN].T + b1) * (X@W12[FFN:2FFN].T + b2).  K=256.
// One wave: one M-tile x two N-tiles x {W1,W2} = 4 WMMAs sharing one A fragment.
__global__ void gemm_swiglu_kernel(const float* __restrict__ X, const float* __restrict__ W12,
                                   const float* __restrict__ b12, float* __restrict__ hidden,
                                   int K) {
  const int groups = FFNQ >> 5;                // 32 pairs of tiles
  const int total  = 3 * groups;               // 96 waves
  const int wave   = blockIdx.x * (blockDim.x >> 5) + (threadIdx.x >> 5);
  if (wave >= total) return;
  const int lane = threadIdx.x & 31;
  const int g    = lane >> 4;
  const int rr   = lane & 15;
  const int mt   = wave / groups;
  const int ng   = wave % groups;
  const float* xrow = X + (size_t)(mt * 16 + rr) * K;
  const float* wa0  = W12 + (size_t)(ng * 32 + rr) * K;          // x1 tile pair
  const float* wa1  = wa0 + (size_t)16 * K;
  const float* wb0  = W12 + (size_t)(FFNQ + ng * 32 + rr) * K;   // x2 tile pair
  const float* wb1  = wb0 + (size_t)16 * K;
  v8f acc1a = {}, acc1b = {}, acc2a = {}, acc2b = {};
  for (int k0 = 0; k0 < K; k0 += 32) {
    __builtin_prefetch(wa0 + k0 + 64, 0, 0);
    v16bf a   = load_frag_a(xrow, k0, g);
    v16bf b1a = load_frag_b(wa0, k0, g);
    v16bf b1b = load_frag_b(wa1, k0, g);
    v16bf b2a = load_frag_b(wb0, k0, g);
    v16bf b2b = load_frag_b(wb1, k0, g);
    acc1a = __builtin_amdgcn_wmma_f32_16x16x32_bf16(false, a, false, b1a, (short)0, acc1a, false, false);
    acc1b = __builtin_amdgcn_wmma_f32_16x16x32_bf16(false, a, false, b1b, (short)0, acc1b, false, false);
    acc2a = __builtin_amdgcn_wmma_f32_16x16x32_bf16(false, a, false, b2a, (short)0, acc2a, false, false);
    acc2b = __builtin_amdgcn_wmma_f32_16x16x32_bf16(false, a, false, b2b, (short)0, acc2b, false, false);
  }
  const int n0 = ng * 32 + rr;
  const int n1 = n0 + 16;
  const float bv1a = b12[n0],        bv1b = b12[n1];
  const float bv2a = b12[FFNQ + n0], bv2b = b12[FFNQ + n1];
#pragma unroll
  for (int r = 0; r < 8; ++r) {
    const int m = (mt << 4) + r + (g << 3);
    const float x1a = acc1a[r] + bv1a, x1b = acc1b[r] + bv1b;
    const float x2a = acc2a[r] + bv2a, x2b = acc2b[r] + bv2b;
    hidden[(size_t)m * FFNQ + n0] = (x1a / (1.0f + __expf(-x1a))) * x2a;
    hidden[(size_t)m * FFNQ + n1] = (x1b / (1.0f + __expf(-x1b))) * x2b;
  }
}

// LayerNorm: one block per row.
__global__ void ln_kernel(const float* __restrict__ x, const float* __restrict__ gam,
                          const float* __restrict__ bet, float* __restrict__ out, int K) {
  __shared__ float s1[256], s2[256];
  const int row = blockIdx.x;
  const float* xr = x + (size_t)row * K;
  float* orow = out + (size_t)row * K;
  float s = 0.0f, sq = 0.0f;
  for (int i = threadIdx.x; i < K; i += blockDim.x) { float v = xr[i]; s += v; sq += v * v; }
  s1[threadIdx.x] = s; s2[threadIdx.x] = sq;
  __syncthreads();
  for (int st = blockDim.x >> 1; st > 0; st >>= 1) {
    if ((int)threadIdx.x < st) { s1[threadIdx.x] += s1[threadIdx.x + st]; s2[threadIdx.x] += s2[threadIdx.x + st]; }
    __syncthreads();
  }
  const float mean = s1[0] / (float)K;
  const float var  = s2[0] / (float)K - mean * mean;
  const float inv  = rsqrtf(var + 1e-5f);
  for (int i = threadIdx.x; i < K; i += blockDim.x)
    orow[i] = (xr[i] - mean) * inv * gam[i] + bet[i];
}

// RoPE: dst = src*cos + rotate_half(src)*sin over [B,L,D]
__global__ void rope_kernel(const float* __restrict__ src, float* __restrict__ dst,
                            const float* __restrict__ cosc, const float* __restrict__ sinc) {
  const int idx = blockIdx.x * blockDim.x + threadIdx.x;
  if (idx >= BQ * LQ * DQ) return;
  const int d = idx % DQ;
  const int l = (idx / DQ) % LQ;
  const float v = src[idx];
  const float other = (d < DQ / 2) ? -src[idx + DQ / 2] : src[idx - DQ / 2];
  dst[idx] = v * cosc[l * DQ + d] + other * sinc[l * DQ + d];
}

// Causal MHA on qkv[48,768]: one thread per (b,h,q) row (192 threads; compute is negligible)
__global__ void attn_kernel(const float* __restrict__ qkv, float* __restrict__ o) {
  const int t = threadIdx.x;
  if (t >= BQ * NHQ * LQ) return;
  const int b = t / (NHQ * LQ);
  const int h = (t / LQ) % NHQ;
  const int q = t % LQ;
  const float* qrow = qkv + (size_t)(b * LQ + q) * (3 * DQ) + h * HDQ;
  float s[LQ];
  float mx = -1e30f;
  for (int j = 0; j <= q; ++j) {
    const float* krow = qkv + (size_t)(b * LQ + j) * (3 * DQ) + DQ + h * HDQ;
    float acc = 0.0f;
#pragma unroll
    for (int d = 0; d < HDQ; ++d) acc += qrow[d] * krow[d];
    acc *= 0.125f;                         // 1/sqrt(64)
    s[j] = acc;
    mx = fmaxf(mx, acc);
  }
  float denom = 0.0f;
  for (int j = 0; j <= q; ++j) { s[j] = __expf(s[j] - mx); denom += s[j]; }
  const float inv = 1.0f / denom;
  float oacc[HDQ];
#pragma unroll
  for (int d = 0; d < HDQ; ++d) oacc[d] = 0.0f;
  for (int j = 0; j <= q; ++j) {
    const float* vrow = qkv + (size_t)(b * LQ + j) * (3 * DQ) + 2 * DQ + h * HDQ;
    const float p = s[j] * inv;
#pragma unroll
    for (int d = 0; d < HDQ; ++d) oacc[d] += p * vrow[d];
  }
  float* orow = o + (size_t)(b * LQ + q) * DQ + h * HDQ;
#pragma unroll
  for (int d = 0; d < HDQ; ++d) orow[d] = oacc[d];
}

// Per-position classifier + entropy gate + halting head. One thread per (b,l).
__global__ void heads_kernel(const float* __restrict__ z, const float* __restrict__ cls_w,
                             const float* __restrict__ cls_b, const float* __restrict__ eg_w1,
                             const float* __restrict__ eg_b1, const float* __restrict__ eg_w2,
                             const float* __restrict__ eg_b2, const float* __restrict__ halt_w,
                             const float* __restrict__ halt_b, float* __restrict__ halt_out,
                             float* __restrict__ cls_out) {
  const int t = threadIdx.x;
  if (t >= MQ) return;
  const int l = t % LQ;
  const float* zr = z + (size_t)t * DQ;
  // classifier logits (C=2)
  float c0 = cls_b[l * CQ + 0], c1 = cls_b[l * CQ + 1];
  const float* w0 = cls_w + (size_t)(l * CQ + 0) * DQ;
  const float* w1 = w0 + DQ;
  for (int d = 0; d < DQ; ++d) { c0 += zr[d] * w0[d]; c1 += zr[d] * w1[d]; }
  cls_out[t * CQ + 0] = c0; cls_out[t * CQ + 1] = c1;
  // entropy of softmax
  const float mx = fmaxf(c0, c1);
  const float e0 = __expf(c0 - mx), e1 = __expf(c1 - mx);
  float p0 = e0 / (e0 + e1), p1 = e1 / (e0 + e1);
  p0 = fminf(fmaxf(p0, 1e-10f), 1.0f);
  p1 = fminf(fmaxf(p1, 1e-10f), 1.0f);
  const float ent  = -(p0 * __logf(p0) + p1 * __logf(p1));
  const float entn = fminf(fmaxf(ent / 0.69314718056f, 0.0f), 1.0f);
  // entropy gate MLP (D->8->1)
  float hj[8];
#pragma unroll
  for (int j = 0; j < 8; ++j) hj[j] = eg_b1[l * 8 + j];
  for (int d = 0; d < DQ; ++d) {
    const float zv = zr[d];
    const float* wr = eg_w1 + (size_t)(l * DQ + d) * 8;
#pragma unroll
    for (int j = 0; j < 8; ++j) hj[j] += zv * wr[j];
  }
  float gsum = eg_b2[l];
#pragma unroll
  for (int j = 0; j < 8; ++j) gsum += tanhf(hj[j]) * eg_w2[l * 8 + j];
  const float gate  = 1.0f / (1.0f + __expf(-gsum));
  const float gated = entn * gate;
  // halting logit over [z, gated]
  const float* hw = halt_w + (size_t)l * (DQ + 1);
  float hl = halt_b[l] + gated * hw[DQ];
  for (int d = 0; d < DQ; ++d) hl += zr[d] * hw[d];
  halt_out[t] = hl;
}

static inline int gemm_blocks(int N) { return (3 * (N / 32) + 7) / 8; }

extern "C" void kernel_launch(void* const* d_in, const int* in_sizes, int n_in,
                              void* d_out, int out_size, void* d_ws, size_t ws_size,
                              hipStream_t stream) {
  const float* teacher  = (const float*)d_in[0];
  const float* in_ln_g  = (const float*)d_in[1];
  const float* in_ln_b  = (const float*)d_in[2];
  const float* proj_w   = (const float*)d_in[3];
  const float* proj_b   = (const float*)d_in[4];
  const float* cos_c    = (const float*)d_in[5];
  const float* sin_c    = (const float*)d_in[6];
  const float* pre_ln_g = (const float*)d_in[7];
  const float* pre_ln_b = (const float*)d_in[8];
  const float* l_ln1_g  = (const float*)d_in[9];
  const float* l_ln1_b  = (const float*)d_in[10];
  const float* l_inw    = (const float*)d_in[11];
  const float* l_inb    = (const float*)d_in[12];
  const float* l_outw   = (const float*)d_in[13];
  const float* l_outb   = (const float*)d_in[14];
  const float* l_ln2_g  = (const float*)d_in[15];
  const float* l_ln2_b  = (const float*)d_in[16];
  const float* l_w12    = (const float*)d_in[17];
  const float* l_b12    = (const float*)d_in[18];
  const float* l_w3     = (const float*)d_in[19];
  const float* l_b3     = (const float*)d_in[20];
  const float* post_ln_g= (const float*)d_in[21];
  const float* post_ln_b= (const float*)d_in[22];
  const float* eg_w1    = (const float*)d_in[23];
  const float* eg_b1    = (const float*)d_in[24];
  const float* eg_w2    = (const float*)d_in[25];
  const float* eg_b2    = (const float*)d_in[26];
  const float* halt_w   = (const float*)d_in[27];
  const float* halt_b   = (const float*)d_in[28];
  const float* cls_w    = (const float*)d_in[29];
  const float* cls_b    = (const float*)d_in[30];

  float* out = (float*)d_out;
  float* halt_out = out;                 // [48]
  float* cls_out  = out + MQ;            // [96]
  float* z_out    = out + MQ + MQ * CQ;  // [48*256] (post-LN z)

  // workspace carve-up (floats)
  float* ws    = (float*)d_ws;
  float* xln   = ws;                      // 48*1024
  float* xproj = xln   + MQ * DTQ;        // 48*256
  float* xrope = xproj + MQ * DQ;         // 48*256
  float* zbuf  = xrope + MQ * DQ;         // 48*256
  float* xn    = zbuf  + MQ * DQ;         // 48*256
  float* qkv   = xn    + MQ * DQ;         // 48*768
  float* attno = qkv   + MQ * 3 * DQ;     // 48*256
  float* hidden= attno + MQ * DQ;         // 48*1024

  // 1) input LN over D_T, 2) projection GEMM, 3) RoPE, 4) pre-LN
  ln_kernel<<<MQ, 256, 0, stream>>>(teacher, in_ln_g, in_ln_b, xln, DTQ);
  gemm_wmma_kernel<true, false><<<gemm_blocks(DQ), 256, 0, stream>>>(
      xln, proj_w, proj_b, nullptr, xproj, DQ, DTQ);
  rope_kernel<<<(MQ * DQ + 255) / 256, 256, 0, stream>>>(xproj, xrope, cos_c, sin_c);
  ln_kernel<<<MQ, 256, 0, stream>>>(xrope, pre_ln_g, pre_ln_b, zbuf, DQ);

  for (int i = 0; i < NLQ; ++i) {
    const float* ln1g = l_ln1_g + (size_t)i * DQ;
    const float* ln1b = l_ln1_b + (size_t)i * DQ;
    const float* inw  = l_inw  + (size_t)i * 3 * DQ * DQ;
    const float* inb  = l_inb  + (size_t)i * 3 * DQ;
    const float* outw = l_outw + (size_t)i * DQ * DQ;
    const float* outb = l_outb + (size_t)i * DQ;
    const float* ln2g = l_ln2_g + (size_t)i * DQ;
    const float* ln2b = l_ln2_b + (size_t)i * DQ;
    const float* w12  = l_w12  + (size_t)i * 2 * FFNQ * DQ;
    const float* b12  = l_b12  + (size_t)i * 2 * FFNQ;
    const float* w3   = l_w3   + (size_t)i * DQ * FFNQ;
    const float* b3   = l_b3   + (size_t)i * DQ;

    ln_kernel<<<MQ, 256, 0, stream>>>(zbuf, ln1g, ln1b, xn, DQ);
    gemm_wmma_kernel<true, false><<<gemm_blocks(3 * DQ), 256, 0, stream>>>(
        xn, inw, inb, nullptr, qkv, 3 * DQ, DQ);
    attn_kernel<<<1, 192, 0, stream>>>(qkv, attno);
    gemm_wmma_kernel<true, true><<<gemm_blocks(DQ), 256, 0, stream>>>(
        attno, outw, outb, zbuf, zbuf, DQ, DQ);
    ln_kernel<<<MQ, 256, 0, stream>>>(zbuf, ln2g, ln2b, xn, DQ);
    gemm_swiglu_kernel<<<(3 * (FFNQ / 32) + 7) / 8, 256, 0, stream>>>(xn, w12, b12, hidden, DQ);
    gemm_wmma_kernel<true, true><<<gemm_blocks(DQ), 256, 0, stream>>>(
        hidden, w3, b3, zbuf, zbuf, DQ, FFNQ);
  }

  // post-LN straight into d_out's z section, then heads
  ln_kernel<<<MQ, 256, 0, stream>>>(zbuf, post_ln_g, post_ln_b, z_out, DQ);
  heads_kernel<<<1, 64, 0, stream>>>(z_out, cls_w, cls_b, eg_w1, eg_b1, eg_w2, eg_b2,
                                     halt_w, halt_b, halt_out, cls_out);
}
